// FCMT_58909771432345
// MI455X (gfx1250) — matrix operand order repaired and checked by reading, compiled, not run on hardware
//
#include <hip/hip_runtime.h>
#include <hip/hip_bf16.h>
#include <math.h>

// ---------------------------------------------------------------------------
// Model constants (from reference)
// ---------------------------------------------------------------------------
#define PN   36      // patches
#define DD   1024    // model dim
#define DIN  3072    // input dim
#define HN   8       // heads
#define DH   128     // head dim
#define DFF  2048    // ffn dim
#define LAYERS 6
#define LN_EPS 1e-5f

typedef __attribute__((ext_vector_type(16))) __bf16 v16bf;
typedef __attribute__((ext_vector_type(8)))  __bf16 v8bf;
typedef __attribute__((ext_vector_type(8)))  float  v8f;

__device__ __forceinline__ unsigned short f2bf(float f) {
    unsigned int u = __float_as_uint(f);
    unsigned int r = u + 0x7FFFu + ((u >> 16) & 1u);   // round-to-nearest-even
    return (unsigned short)(r >> 16);
}
__device__ __forceinline__ float geluf(float x) {
    return 0.5f * x * (1.0f + erff(x * 0.7071067811865475f));
}

// ---------------------------------------------------------------------------
// fp32 -> bf16 elementwise convert (weights, once per launch)
// ---------------------------------------------------------------------------
__global__ void cvt_bf16_kernel(const float* __restrict__ in,
                                unsigned short* __restrict__ out, int n) {
    int i = blockIdx.x * blockDim.x + threadIdx.x;
    if (i < n) out[i] = f2bf(in[i]);
}

// ---------------------------------------------------------------------------
// LayerNorm: one block (256 thr = 8 waves) per row.  Emits bf16 (GEMM input)
// or fp32 (final output).
// ---------------------------------------------------------------------------
__global__ __launch_bounds__(256) void ln_kernel(
    const float* __restrict__ in, const float* __restrict__ g,
    const float* __restrict__ b, int width,
    unsigned short* __restrict__ outBf, float* __restrict__ outF)
{
    __shared__ float sSum[256];
    __shared__ float sSq[256];
    const int row = blockIdx.x;
    const float* x = in + (long)row * width;
    float s = 0.f, q = 0.f;
    for (int c = threadIdx.x; c < width; c += 256) { float v = x[c]; s += v; q += v * v; }
    sSum[threadIdx.x] = s; sSq[threadIdx.x] = q;
    __syncthreads();
    for (int st = 128; st > 0; st >>= 1) {
        if (threadIdx.x < st) {
            sSum[threadIdx.x] += sSum[threadIdx.x + st];
            sSq[threadIdx.x]  += sSq[threadIdx.x + st];
        }
        __syncthreads();
    }
    const float mean = sSum[0] / (float)width;
    const float var  = sSq[0] / (float)width - mean * mean;
    const float inv  = rsqrtf(var + LN_EPS);
    for (int c = threadIdx.x; c < width; c += 256) {
        float v = (x[c] - mean) * inv * g[c] + b[c];
        if (outBf) outBf[(long)row * width + c] = f2bf(v);
        else       outF [(long)row * width + c] = v;
    }
}

// ---------------------------------------------------------------------------
// bf16 WMMA GEMM:  C[M,N] = act(A[M,K] @ W[N,K]^T + bias) (+ resid)
//
// 128-thr block = 4 waves; wave w owns a 32(M) x 64(N) macro-tile as a 2x4
// grid of 16x16 f32 accumulators (block covers 128 x 64).  K stepped by 32.
// Per K-step: 4 A b128-loads + 8 B b128-loads feed 8 WMMAs (1.5 loads/WMMA).
// A lane layout 16x32, K contiguous (same as B): lanes 0-15 hold K {0..15},
// lanes 16-31 hold K {16..31}; B lanes hold one column, K contiguous.
// ---------------------------------------------------------------------------
__global__ __launch_bounds__(128) void gemm_wmma_kernel(
    const unsigned short* __restrict__ A, const unsigned short* __restrict__ W,
    const float* __restrict__ bias, const float* __restrict__ resid,
    float* __restrict__ outF, unsigned short* __restrict__ outBf,
    int M, int N, int K, int act)
{
    const int wave = threadIdx.x >> 5;
    const int lane = threadIdx.x & 31;
    const int half = lane >> 4;        // 0: lanes 0-15, 1: lanes 16-31
    const int lrow = lane & 15;
    const int m0 = blockIdx.y * 128 + wave * 32;
    const int n0 = blockIdx.x * 64;

    v8f acc[2][4] = {};
    const unsigned short* arow0 = A + (long)(m0 + lrow) * K;        // rows m0..m0+15
    const unsigned short* arow1 = arow0 + (long)16 * K;             // rows m0+16..m0+31
    const unsigned short* wrow  = W + (long)(n0 + lrow) * K;

    for (int k0 = 0; k0 < K; k0 += 32) {
        // gfx1250 prefetch of the next streamed A chunk (global_prefetch_b8)
        if (k0 + 32 < K) {
            __builtin_prefetch(arow0 + k0 + 32, 0, 1);
            __builtin_prefetch(arow1 + k0 + 32, 0, 1);
        }

        // Two A fragments 16x32 (this wave's 32 rows)
        v16bf a[2];
        {
            v8bf lo0 = *(const v8bf*)(arow0 + k0 + half * 16);
            v8bf hi0 = *(const v8bf*)(arow0 + k0 + 8 + half * 16);
            v8bf lo1 = *(const v8bf*)(arow1 + k0 + half * 16);
            v8bf hi1 = *(const v8bf*)(arow1 + k0 + 8 + half * 16);
            #pragma unroll
            for (int i = 0; i < 8; ++i) {
                a[0][i] = lo0[i]; a[0][i + 8] = hi0[i];
                a[1][i] = lo1[i]; a[1][i + 8] = hi1[i];
            }
        }

        // Four B fragments 32x16, each reused by both A fragments
        #pragma unroll
        for (int j = 0; j < 4; ++j) {
            v16bf bt = *(const v16bf*)(wrow + (long)j * 16 * K + k0 + half * 16);
            acc[0][j] = __builtin_amdgcn_wmma_f32_16x16x32_bf16(
                false, a[0], false, bt, (short)0, acc[0][j], false, false);
            acc[1][j] = __builtin_amdgcn_wmma_f32_16x16x32_bf16(
                false, a[1], false, bt, (short)0, acc[1][j], false, false);
        }
    }

    // Epilogue.  C/D layout: VGPR r -> (M = r + half*8, N = lrow) within tile.
    #pragma unroll
    for (int mi = 0; mi < 2; ++mi) {
        #pragma unroll
        for (int j = 0; j < 4; ++j) {
            const int col = n0 + j * 16 + lrow;
            const float bv = bias ? bias[col] : 0.f;
            #pragma unroll
            for (int r = 0; r < 8; ++r) {
                const int row = m0 + mi * 16 + half * 8 + r;
                float v = acc[mi][j][r] + bv;
                if (act == 1)      v = geluf(v);
                else if (act == 2) v = fmaxf(v, 0.f);
                const long idx = (long)row * N + col;
                if (resid) v += resid[idx];
                if (outBf) outBf[idx] = f2bf(v);
                else       outF[idx]  = v;
            }
        }
    }
}

// ---------------------------------------------------------------------------
// Leave-one-out attention maps.  One 64-thread block per batch element.
// h1 = gelu(xr @ w1^T + b1); full = h1 @ w2 + b2; maps = sigmoid(full[loo]/T)
// ---------------------------------------------------------------------------
__global__ __launch_bounds__(64) void maps_kernel(
    const float* __restrict__ xr, const float* __restrict__ w1,
    const float* __restrict__ b1, const float* __restrict__ w2,
    const float* __restrict__ b2, const float* __restrict__ temp,
    float* __restrict__ maps)
{
    __shared__ float red[64];
    __shared__ float fullS[PN];
    const int bidx = blockIdx.x;
    const int o = threadIdx.x;
    const float* w1o = w1 + (long)o * DD;
    const float w2o = w2[o];
    for (int p = 0; p < PN; ++p) {
        const float* xp = xr + ((long)bidx * PN + p) * DD;
        float dot = 0.f;
        for (int c = 0; c < DD; ++c) dot += xp[c] * w1o[c];
        red[o] = geluf(dot + b1[o]) * w2o;
        __syncthreads();
        for (int st = 32; st > 0; st >>= 1) {
            if (o < st) red[o] += red[o + st];
            __syncthreads();
        }
        if (o == 0) fullS[p] = red[0] + b2[0];
        __syncthreads();
    }
    const float invT = 1.0f / temp[0];
    for (int e = o; e < PN * (PN - 1); e += 64) {
        const int i = e / (PN - 1), jj = e % (PN - 1);
        const int j = (jj < i) ? jj : jj + 1;
        const float v = fullS[j] * invT;
        maps[(long)bidx * PN * (PN - 1) + e] = 1.0f / (1.0f + expf(-v));
    }
}

// ---------------------------------------------------------------------------
// MHA for P=36 (tiny: ~0.8% of total FLOPs) — fp32 VALU per (b,h) block.
// Reads packed qkv (M,3072), writes bf16 O ready for the Wo GEMM.
// ---------------------------------------------------------------------------
__global__ __launch_bounds__(128) void attn_kernel(
    const float* __restrict__ qkv, unsigned short* __restrict__ obf)
{
    __shared__ float S[PN][PN];
    const int b = blockIdx.x >> 3;
    const int h = blockIdx.x & 7;
    const int t = threadIdx.x;
    const float scale = 0.08838834764831845f;   // 1/sqrt(128)
    const long base = (long)b * PN * (3 * DD);

    for (int e = t; e < PN * PN; e += 128) {
        const int qi = e / PN, ki = e % PN;
        const float* qp = qkv + base + (long)qi * (3 * DD) + h * DH;
        const float* kp = qkv + base + (long)ki * (3 * DD) + DD + h * DH;
        float dot = 0.f;
        for (int d = 0; d < DH; ++d) dot += qp[d] * kp[d];
        S[qi][ki] = dot * scale;
    }
    __syncthreads();
    if (t < PN) {
        float mx = -1e30f;
        for (int k = 0; k < PN; ++k) mx = fmaxf(mx, S[t][k]);
        float sum = 0.f;
        for (int k = 0; k < PN; ++k) { float e_ = expf(S[t][k] - mx); S[t][k] = e_; sum += e_; }
        const float inv = 1.0f / sum;
        for (int k = 0; k < PN; ++k) S[t][k] *= inv;
    }
    __syncthreads();
    for (int e = t; e < PN * DH; e += 128) {
        const int p = e >> 7, d = e & 127;
        const float* vp = qkv + base + 2 * DD + h * DH + d;
        float o = 0.f;
        for (int k = 0; k < PN; ++k) o += S[p][k] * vp[(long)k * (3 * DD)];
        obf[((long)b * PN + p) * DD + h * DH + d] = f2bf(o);
    }
}

// ---------------------------------------------------------------------------
// Launcher
// ---------------------------------------------------------------------------
extern "C" void kernel_launch(void* const* d_in, const int* in_sizes, int n_in,
                              void* d_out, int out_size, void* d_ws, size_t ws_size,
                              hipStream_t stream) {
    const float* x     = (const float*)d_in[0];
    const float* ln0_g = (const float*)d_in[1];
    const float* ln0_b = (const float*)d_in[2];
    const float* Wred  = (const float*)d_in[3];
    const float* bred  = (const float*)d_in[4];
    const float* h0_w1 = (const float*)d_in[5];
    const float* h0_b1 = (const float*)d_in[6];
    const float* h0_w2 = (const float*)d_in[7];
    const float* h0_b2 = (const float*)d_in[8];
    const float* temp  = (const float*)d_in[9];
    const float* Wqkv  = (const float*)d_in[10];
    const float* bqkv  = (const float*)d_in[11];
    const float* Wo    = (const float*)d_in[12];
    const float* bo    = (const float*)d_in[13];
    const float* ln1_g = (const float*)d_in[14];
    const float* ln1_b = (const float*)d_in[15];
    const float* W1    = (const float*)d_in[16];
    const float* b1    = (const float*)d_in[17];
    const float* W2    = (const float*)d_in[18];
    const float* b2    = (const float*)d_in[19];
    const float* ln2_g = (const float*)d_in[20];
    const float* ln2_b = (const float*)d_in[21];
    const float* lnf_g = (const float*)d_in[22];
    const float* lnf_b = (const float*)d_in[23];

    const int B = 128;
    const int M = B * PN;                 // 4608 tokens (36 x 128)

    // ---- workspace carve-up (~215 MB) ----
    char* ws = (char*)d_ws;
    size_t off = 0;
    auto give = [&](size_t bytes) -> void* {
        void* p = ws + off;
        off += (bytes + 255) & ~(size_t)255;
        return p;
    };
    unsigned short* wredb = (unsigned short*)give((size_t)DD * DIN * 2);
    unsigned short* wqkvb = (unsigned short*)give((size_t)LAYERS * 3 * DD * DD * 2);
    unsigned short* wob   = (unsigned short*)give((size_t)LAYERS * DD * DD * 2);
    unsigned short* w1b   = (unsigned short*)give((size_t)LAYERS * DFF * DD * 2);
    unsigned short* w2b   = (unsigned short*)give((size_t)LAYERS * DD * DFF * 2);
    float*          y     = (float*)give((size_t)M * DD * 4);           // xr, then residual stream
    float*          qkvF  = (float*)give((size_t)M * 3 * DD * 4);
    unsigned short* actbf = (unsigned short*)give((size_t)M * DIN * 2); // bf16 staging

    float* outF  = (float*)d_out;
    float* mapsF = outF + (size_t)M * DD;

    auto cvt = [&](const float* src, unsigned short* dst, long n) {
        cvt_bf16_kernel<<<dim3((unsigned)((n + 255) / 256)), dim3(256), 0, stream>>>(src, dst, (int)n);
    };
    auto gemm = [&](const unsigned short* A, const unsigned short* Wb, const float* bias,
                    const float* resid, float* oF, unsigned short* oBf,
                    int Mm, int Nn, int Kk, int act) {
        gemm_wmma_kernel<<<dim3(Nn / 64, Mm / 128), dim3(128), 0, stream>>>(
            A, Wb, bias, resid, oF, oBf, Mm, Nn, Kk, act);
    };
    auto ln = [&](const float* in, const float* g, const float* b, int width,
                  unsigned short* oBf, float* oF) {
        ln_kernel<<<dim3(M), dim3(256), 0, stream>>>(in, g, b, width, oBf, oF);
    };

    // ---- 1) weights -> bf16 (amortized over whole net; all fit in 192MB L2) ----
    cvt(Wred, wredb, (long)DD * DIN);
    cvt(Wqkv, wqkvb, (long)LAYERS * 3 * DD * DD);
    cvt(Wo,   wob,   (long)LAYERS * DD * DD);
    cvt(W1,   w1b,   (long)LAYERS * DFF * DD);
    cvt(W2,   w2b,   (long)LAYERS * DD * DFF);

    // ---- 2) dim_reduce: xr = gelu(LN0(x) @ Wred^T + bred) ----
    ln(x, ln0_g, ln0_b, DIN, actbf, nullptr);
    gemm(actbf, wredb, bred, nullptr, y, nullptr, M, DD, DIN, /*gelu*/1);

    // ---- 3) leave-one-out attention maps (reads xr before layers mutate it) ----
    maps_kernel<<<dim3(B), dim3(64), 0, stream>>>(y, h0_w1, h0_b1, h0_w2, h0_b2, temp, mapsF);

    // ---- 4) transformer encoder (pre-norm), y updated in place ----
    for (int l = 0; l < LAYERS; ++l) {
        const unsigned short* wq = wqkvb + (size_t)l * 3 * DD * DD;
        const unsigned short* wo = wob   + (size_t)l * DD * DD;
        const unsigned short* wu = w1b   + (size_t)l * DFF * DD;
        const unsigned short* wd = w2b   + (size_t)l * DD * DFF;

        // attention sub-block
        ln(y, ln1_g + (size_t)l * DD, ln1_b + (size_t)l * DD, DD, actbf, nullptr);
        gemm(actbf, wq, bqkv + (size_t)l * 3 * DD, nullptr, qkvF, nullptr, M, 3 * DD, DD, 0);
        attn_kernel<<<dim3(B * HN), dim3(128), 0, stream>>>(qkvF, actbf);   // O -> bf16
        gemm(actbf, wo, bo + (size_t)l * DD, /*resid=*/y, y, nullptr, M, DD, DD, 0);

        // FFN sub-block
        ln(y, ln2_g + (size_t)l * DD, ln2_b + (size_t)l * DD, DD, actbf, nullptr);
        unsigned short* ffnbf = actbf + (size_t)M * DD;   // disjoint region of staging buf
        gemm(actbf, wu, b1 + (size_t)l * DFF, nullptr, nullptr, ffnbf, M, DFF, DD, /*relu*/2);
        gemm(ffnbf, wd, b2 + (size_t)l * DD, /*resid=*/y, y, nullptr, M, DD, DFF, 0);
    }

    // ---- 5) final LN -> d_out ----
    ln(y, lnf_g, lnf_b, DD, nullptr, outF);
}
